// DeltaNetEnhancedMoE_10557029613743
// MI455X (gfx1250) — matrix operand order repaired and checked by reading, not compiled, not run on hardware
//
#include <hip/hip_runtime.h>
#include <hip/hip_bf16.h>
#include <cstdint>
#include <cstddef>

// ---------------- problem constants ----------------
#define T_TOK 8192          // B*N tokens
#define C_DIM 2048
#define NE 8
#define CAP 1280            // capacity per expert
#define I_DIM 5461          // SwiGLU intermediate
#define IPAD 5472           // padded to multiple of 32 (K of down GEMM)
#define GIN_K 2054          // C + 6 stats
#define GIN_LD 2080         // padded to multiple of 32 (K of gate GEMM)
#define H_DIM 1024          // gate hidden
#define MT_CAP (CAP / 16)   // 80 m-tiles per expert
#define LSTR 40             // LDS row stride in elements (80B: pads banks, keeps 16B align)

typedef __attribute__((ext_vector_type(8)))  __bf16 v8bf;
typedef __attribute__((ext_vector_type(16))) __bf16 v16bf;
typedef __attribute__((ext_vector_type(8)))  float  v8f;
typedef __attribute__((ext_vector_type(4)))  int    v4i;

// Prefer the gfx1250 async global->LDS copy path (ASYNCcnt) when the toolchain
// exposes the builtins; otherwise fall back to load + ds_store.
#if defined(__has_builtin)
#if __has_builtin(__builtin_amdgcn_global_load_async_to_lds_b128) && \
    __has_builtin(__builtin_amdgcn_s_wait_asynccnt)
#define USE_ASYNC_LDS 1
#endif
#endif

__device__ __forceinline__ void stage16(const __bf16* g, __bf16* l) {
#ifdef USE_ASYNC_LDS
  // signature (from hipcc diagnostic): (global int4*, local int4*, imm offset, imm cpol)
  __builtin_amdgcn_global_load_async_to_lds_b128(
      (__attribute__((address_space(1))) v4i*)g,
      (__attribute__((address_space(3))) v4i*)l, 0, 0);
#else
  *(v8bf*)l = *(const v8bf*)g;
#endif
}
__device__ __forceinline__ void wait_stage() {
#ifdef USE_ASYNC_LDS
  __builtin_amdgcn_s_wait_asynccnt(0);
#endif
}

// D = A(16x32 bf16) * B(32x16 bf16) + C(16x16 f32)
__device__ __forceinline__ v8f wmma_bf16(v16bf a, v16bf b, v8f c) {
  return __builtin_amdgcn_wmma_f32_16x16x32_bf16(false, a, false, b, (short)0, c,
                                                 false, false);
}

// A-matrix 16x32 bf16 layout (matches B): lanes 0-15 hold K=[0..16),
// lanes 16-31 hold K=[16..32) of row M=lane%16 -> two 16B loads.
__device__ __forceinline__ v16bf load_a16x32(const __bf16* row, int kb, int aoff) {
  v8bf lo = *(const v8bf*)(row + kb + 2 * aoff);
  v8bf hi = *(const v8bf*)(row + kb + 2 * aoff + 8);
  return __builtin_shufflevector(lo, hi, 0, 1, 2, 3, 4, 5, 6, 7,
                                 8, 9, 10, 11, 12, 13, 14, 15);
}

// 16 contiguous bf16 (32B) as two 16B chunks (works with 80B LDS row stride)
__device__ __forceinline__ v16bf load_frag32(const __bf16* p) {
  v8bf lo = *(const v8bf*)p;
  v8bf hi = *(const v8bf*)(p + 8);
  return __builtin_shufflevector(lo, hi, 0, 1, 2, 3, 4, 5, 6, 7,
                                 8, 9, 10, 11, 12, 13, 14, 15);
}

__device__ __forceinline__ float gelu_exact(float v) {
  return 0.5f * v * (1.0f + erff(v * 0.70710678118654752f));
}

// ---------------- 1) token stats + bf16 pack of gate input ----------------
__global__ void stats_gin_kernel(const float* __restrict__ x, __bf16* __restrict__ gin) {
  int wid = threadIdx.x >> 5, lane = threadIdx.x & 31;
  int t = blockIdx.x * 8 + wid;
  const float* row = x + (size_t)t * C_DIM;
  __bf16* grow = gin + (size_t)t * GIN_LD;
  float s = 0.f, s2 = 0.f, mn = 3.4e38f, mx = -3.4e38f, cnt = 0.f;
  for (int j = lane; j < C_DIM; j += 32) {
    float v = row[j];
    grow[j] = (__bf16)v;
    s += v; s2 += v * v;
    mn = fminf(mn, v); mx = fmaxf(mx, v);
    cnt += (fabsf(v) < 1e-6f) ? 1.f : 0.f;
  }
  #pragma unroll
  for (int o = 16; o; o >>= 1) {
    s   += __shfl_xor(s, o, 32);
    s2  += __shfl_xor(s2, o, 32);
    mn   = fminf(mn, __shfl_xor(mn, o, 32));
    mx   = fmaxf(mx, __shfl_xor(mx, o, 32));
    cnt += __shfl_xor(cnt, o, 32);
  }
  if (lane == 0) {
    float mean = s / (float)C_DIM;
    float var = fmaxf(0.f, (s2 - (float)C_DIM * mean * mean) / (float)(C_DIM - 1));
    grow[2048] = (__bf16)mean;
    grow[2049] = (__bf16)sqrtf(var);
    grow[2050] = (__bf16)mn;
    grow[2051] = (__bf16)mx;
    grow[2052] = (__bf16)sqrtf(s2);
    grow[2053] = (__bf16)(cnt / (float)C_DIM);
    for (int j = GIN_K; j < GIN_LD; ++j) grow[j] = (__bf16)0.f;  // zero pad
  }
}

// ---------------- 2) fp32 -> bf16 weight repacks ----------------
__global__ void conv_g1w_kernel(const float* __restrict__ g1w, __bf16* __restrict__ dst) {
  size_t i = (size_t)blockIdx.x * 256 + threadIdx.x;
  if (i >= (size_t)H_DIM * GIN_LD) return;
  int r = (int)(i / GIN_LD), k = (int)(i % GIN_LD);
  float v = (k < GIN_K) ? g1w[(size_t)r * GIN_K + k] : 0.f;
  dst[i] = (__bf16)v;
}

// w1 (E, 2I, C) -> (E, 2*IPAD, C): rows [0,IPAD)=gate, [IPAD,2*IPAD)=up, zero pad rows
__global__ void conv_w1_kernel(const float* __restrict__ w1, __bf16* __restrict__ w1b) {
  const size_t per_e = (size_t)2 * IPAD * C_DIM;
  size_t i = (size_t)blockIdx.x * 256 + threadIdx.x;
  if (i >= (size_t)NE * per_e) return;
  int e = (int)(i / per_e);
  size_t r = i % per_e;
  int row = (int)(r / C_DIM), c = (int)(r % C_DIM);
  int half = row / IPAD, rr = row % IPAD;
  float v = 0.f;
  if (rr < I_DIM)
    v = w1[((size_t)e * 2 * I_DIM + (size_t)half * I_DIM + rr) * C_DIM + c];
  w1b[i] = (__bf16)v;
}

// w2 (E, C, I) -> (E, C, IPAD), zero K-pad
__global__ void conv_w2_kernel(const float* __restrict__ w2, __bf16* __restrict__ w2b) {
  const size_t per_e = (size_t)C_DIM * IPAD;
  size_t i = (size_t)blockIdx.x * 256 + threadIdx.x;
  if (i >= (size_t)NE * per_e) return;
  int e = (int)(i / per_e);
  size_t r = i % per_e;
  int c = (int)(r / IPAD), k = (int)(r % IPAD);
  float v = (k < I_DIM) ? w2[((size_t)e * C_DIM + c) * I_DIM + k] : 0.f;
  w2b[i] = (__bf16)v;
}

// ---------------- 3) gate GEMM1 (WMMA bf16) + GELU ----------------
// wave computes 16 tokens x 64 outputs; K = GIN_LD (65 steps of 32)
__global__ void gate_gemm_kernel(const __bf16* __restrict__ gin,
                                 const __bf16* __restrict__ g1wb,
                                 const float* __restrict__ g1b,
                                 __bf16* __restrict__ h) {
  int gw = blockIdx.x * 8 + (threadIdx.x >> 5);
  int lane = threadIdx.x & 31;
  int mt = gw >> 4;            // 512 token tiles
  int nb = (gw & 15) * 64;     // 16 output chunks of 64
  int nl = lane & 15;
  const __bf16* arow = gin + (size_t)(mt * 16 + nl) * GIN_LD;
  int aoff  = (lane < 16) ? 0 : 8;
  int koffB = (lane < 16) ? 0 : 16;   // B 32x16: lanes 0-15 K=[0..16), 16-31 K=[16..32)
  const __bf16* b0 = g1wb + (size_t)(nb + 0 * 16 + nl) * GIN_LD + koffB;
  const __bf16* b1 = g1wb + (size_t)(nb + 1 * 16 + nl) * GIN_LD + koffB;
  const __bf16* b2 = g1wb + (size_t)(nb + 2 * 16 + nl) * GIN_LD + koffB;
  const __bf16* b3 = g1wb + (size_t)(nb + 3 * 16 + nl) * GIN_LD + koffB;
  v8f a0 = {}, a1 = {}, a2 = {}, a3 = {};
  for (int kb = 0; kb < GIN_LD; kb += 32) {
    v16bf a = load_a16x32(arow, kb, aoff);
    __builtin_prefetch(b0 + kb + 512, 0, 1);
    a0 = wmma_bf16(a, *(const v16bf*)(b0 + kb), a0);
    a1 = wmma_bf16(a, *(const v16bf*)(b1 + kb), a1);
    a2 = wmma_bf16(a, *(const v16bf*)(b2 + kb), a2);
    a3 = wmma_bf16(a, *(const v16bf*)(b3 + kb), a3);
  }
  int hi = (lane < 16) ? 0 : 8;     // C/D: vgpr r holds M=r (lanes 0-15) / M=r+8
  #pragma unroll
  for (int r = 0; r < 8; ++r) {
    size_t row = (size_t)(mt * 16 + r + hi) * H_DIM;
    h[row + nb + 0 * 16 + nl] = (__bf16)gelu_exact(a0[r] + g1b[nb + 0 * 16 + nl]);
    h[row + nb + 1 * 16 + nl] = (__bf16)gelu_exact(a1[r] + g1b[nb + 1 * 16 + nl]);
    h[row + nb + 2 * 16 + nl] = (__bf16)gelu_exact(a2[r] + g1b[nb + 2 * 16 + nl]);
    h[row + nb + 3 * 16 + nl] = (__bf16)gelu_exact(a3[r] + g1b[nb + 3 * 16 + nl]);
  }
}

// ---------------- 4) gate2 logits + softmax + top-2 ----------------
__global__ void gate2_route_kernel(const __bf16* __restrict__ h,
                                   const float* __restrict__ g2w,
                                   int* __restrict__ eid, float* __restrict__ ewt) {
  int t = blockIdx.x * 8 + (threadIdx.x >> 5);
  int lane = threadIdx.x & 31;
  const __bf16* hrow = h + (size_t)t * H_DIM;
  float acc[NE];
  #pragma unroll
  for (int e = 0; e < NE; ++e) acc[e] = 0.f;
  for (int j = lane; j < H_DIM; j += 32) {
    float hv = (float)hrow[j];
    #pragma unroll
    for (int e = 0; e < NE; ++e) acc[e] += hv * g2w[e * H_DIM + j];
  }
  #pragma unroll
  for (int o = 16; o; o >>= 1)
    #pragma unroll
    for (int e = 0; e < NE; ++e) acc[e] += __shfl_xor(acc[e], o, 32);
  if (lane == 0) {
    float m = acc[0];
    #pragma unroll
    for (int e = 1; e < NE; ++e) m = fmaxf(m, acc[e]);
    float p[NE];
    #pragma unroll
    for (int e = 0; e < NE; ++e) p[e] = expf(acc[e] - m);
    int b0 = 0;
    #pragma unroll
    for (int e = 1; e < NE; ++e) if (p[e] > p[b0]) b0 = e;
    int b1 = (b0 == 0) ? 1 : 0;
    #pragma unroll
    for (int e = 0; e < NE; ++e) if (e != b0 && p[e] > p[b1]) b1 = e;
    float sw = p[b0] + p[b1];
    eid[2 * t + 0] = b0; ewt[2 * t + 0] = p[b0] / sw;
    eid[2 * t + 1] = b1; ewt[2 * t + 1] = p[b1] / sw;
  }
}

// ---------------- 5) deterministic per-expert capacity compaction ----------------
__global__ void route_compact_kernel(const int* __restrict__ eid,
                                     const float* __restrict__ ewt,
                                     int* __restrict__ tokidx, float* __restrict__ slotwt,
                                     int* __restrict__ slotpos, int* __restrict__ cnt) {
  int e = blockIdx.x;
  __shared__ int sscan[1024];
  __shared__ int s_base;
  if (threadIdx.x == 0) s_base = 0;
  __syncthreads();
  for (int s0 = 0; s0 < 2 * T_TOK; s0 += 1024) {
    int s = s0 + threadIdx.x;
    int pred = (eid[s] == e) ? 1 : 0;
    sscan[threadIdx.x] = pred;
    __syncthreads();
    for (int off = 1; off < 1024; off <<= 1) {       // inclusive Hillis-Steele
      int add = (threadIdx.x >= off) ? sscan[threadIdx.x - off] : 0;
      __syncthreads();
      sscan[threadIdx.x] += add;
      __syncthreads();
    }
    int base = s_base;
    int pos = base + sscan[threadIdx.x] - 1;
    if (pred) {
      if (pos < CAP) {
        tokidx[e * CAP + pos] = s >> 1;
        slotwt[e * CAP + pos] = ewt[s];
        slotpos[s] = pos;
      } else {
        slotpos[s] = -1;                             // dropped (over capacity)
      }
    }
    int total = sscan[1023];
    __syncthreads();
    if (threadIdx.x == 0) s_base = base + total;
    __syncthreads();
  }
  int c = s_base; if (c > CAP) c = CAP;
  for (int p = c + threadIdx.x; p < CAP; p += 1024) {  // pad: token 0, weight 0
    tokidx[e * CAP + p] = 0;
    slotwt[e * CAP + p] = 0.f;
  }
  if (threadIdx.x == 0) cnt[e] = c;
}

// ---------------- 6) expert up-proj + fused SwiGLU (WMMA bf16) ----------------
// Block = 8 waves sharing one 32-wide intermediate column (gate+up = 64 B-rows);
// B tile (4KB/K-step) double-buffered in LDS via async global->LDS copies.
// Each wave owns one 16-row capacity tile (A gathered per token).
__global__ void expert_up_kernel(const __bf16* __restrict__ gin,
                                 const __bf16* __restrict__ w1b,
                                 const int* __restrict__ tokidx,
                                 __bf16* __restrict__ act) {
  const int NT = IPAD / 32;  // 171
  __shared__ __bf16 lbuf[2][64 * LSTR];
  int b = blockIdx.x;
  int e = b / ((MT_CAP / 8) * NT);
  int rem = b % ((MT_CAP / 8) * NT);
  int mtg = rem / NT, nt = rem % NT;
  int wid = threadIdx.x >> 5, lane = threadIdx.x & 31;
  int mt = mtg * 8 + wid;
  int nl = lane & 15;

  // staging role: thread -> (row 0..63, 16B chunk 0..3)
  int srow = threadIdx.x >> 2;
  int sch  = (threadIdx.x & 3) * 8;
  int growi = (srow < 32) ? (nt * 32 + srow) : (IPAD + nt * 32 + (srow - 32));
  const __bf16* gsrc = w1b + (size_t)e * 2 * IPAD * C_DIM + (size_t)growi * C_DIM + sch;
  __bf16* ldst0 = &lbuf[0][srow * LSTR + sch];
  __bf16* ldst1 = &lbuf[1][srow * LSTR + sch];

  int tok = tokidx[e * CAP + mt * 16 + nl];
  const __bf16* arow = gin + (size_t)tok * GIN_LD;   // gathered token row (bf16)
  int aoff  = (lane < 16) ? 0 : 8;
  int koffB = (lane < 16) ? 0 : 16;

  stage16(gsrc, ldst0);                              // prologue
  wait_stage();
  __syncthreads();

  v8f aG0 = {}, aG1 = {}, aU0 = {}, aU1 = {};
  int buf = 0;
  for (int kb = 0; kb < C_DIM; kb += 32) {
    if (kb + 32 < C_DIM)
      stage16(gsrc + kb + 32, buf ? ldst0 : ldst1);  // fill other buffer
    __builtin_prefetch(gsrc + kb + 256, 0, 1);
    v16bf a = load_a16x32(arow, kb, aoff);
    const __bf16* lb = &lbuf[buf][0];
    v16bf Bg0 = load_frag32(lb + (0  + nl) * LSTR + koffB);
    v16bf Bg1 = load_frag32(lb + (16 + nl) * LSTR + koffB);
    v16bf Bu0 = load_frag32(lb + (32 + nl) * LSTR + koffB);
    v16bf Bu1 = load_frag32(lb + (48 + nl) * LSTR + koffB);
    aG0 = wmma_bf16(a, Bg0, aG0);
    aG1 = wmma_bf16(a, Bg1, aG1);
    aU0 = wmma_bf16(a, Bu0, aU0);
    aU1 = wmma_bf16(a, Bu1, aU1);
    wait_stage();
    __syncthreads();
    buf ^= 1;
  }

  int hi = (lane < 16) ? 0 : 8;
  #pragma unroll
  for (int r = 0; r < 8; ++r) {
    int row = mt * 16 + r + hi;
    __bf16* ar = act + ((size_t)e * CAP + row) * IPAD + nt * 32 + nl;
    float g0 = aG0[r], u0 = aU0[r];
    float g1 = aG1[r], u1 = aU1[r];
    ar[0]  = (__bf16)((g0 / (1.f + expf(-g0))) * u0);   // silu(g)*u
    ar[16] = (__bf16)((g1 / (1.f + expf(-g1))) * u1);
  }
}

// ---------------- 7) expert down-proj (WMMA bf16), gate weight at store ----------------
// Block = 8 waves sharing one 64-wide output chunk (64 B-rows of w2), LDS staged.
__global__ void expert_down_kernel(const __bf16* __restrict__ act,
                                   const __bf16* __restrict__ w2b,
                                   const float* __restrict__ slotwt,
                                   float* __restrict__ eout) {
  const int NT = C_DIM / 64;  // 32
  __shared__ __bf16 lbuf[2][64 * LSTR];
  int b = blockIdx.x;
  int e = b / ((MT_CAP / 8) * NT);
  int rem = b % ((MT_CAP / 8) * NT);
  int mtg = rem / NT, nb = (rem % NT) * 64;
  int wid = threadIdx.x >> 5, lane = threadIdx.x & 31;
  int mt = mtg * 8 + wid;
  int nl = lane & 15;

  int srow = threadIdx.x >> 2;
  int sch  = (threadIdx.x & 3) * 8;
  const __bf16* gsrc = w2b + (size_t)e * C_DIM * IPAD + (size_t)(nb + srow) * IPAD + sch;
  __bf16* ldst0 = &lbuf[0][srow * LSTR + sch];
  __bf16* ldst1 = &lbuf[1][srow * LSTR + sch];

  const __bf16* arow = act + ((size_t)e * CAP + mt * 16 + nl) * IPAD;
  int aoff  = (lane < 16) ? 0 : 8;
  int koffB = (lane < 16) ? 0 : 16;

  stage16(gsrc, ldst0);
  wait_stage();
  __syncthreads();

  v8f a0 = {}, a1 = {}, a2 = {}, a3 = {};
  int buf = 0;
  for (int kb = 0; kb < IPAD; kb += 32) {
    if (kb + 32 < IPAD)
      stage16(gsrc + kb + 32, buf ? ldst0 : ldst1);
    __builtin_prefetch(gsrc + kb + 256, 0, 1);
    v16bf a = load_a16x32(arow, kb, aoff);
    const __bf16* lb = &lbuf[buf][0];
    v16bf B0 = load_frag32(lb + (0  + nl) * LSTR + koffB);
    v16bf B1 = load_frag32(lb + (16 + nl) * LSTR + koffB);
    v16bf B2 = load_frag32(lb + (32 + nl) * LSTR + koffB);
    v16bf B3 = load_frag32(lb + (48 + nl) * LSTR + koffB);
    a0 = wmma_bf16(a, B0, a0);
    a1 = wmma_bf16(a, B1, a1);
    a2 = wmma_bf16(a, B2, a2);
    a3 = wmma_bf16(a, B3, a3);
    wait_stage();
    __syncthreads();
    buf ^= 1;
  }

  int hi = (lane < 16) ? 0 : 8;
  #pragma unroll
  for (int r = 0; r < 8; ++r) {
    int row = mt * 16 + r + hi;
    float w = slotwt[e * CAP + row];                 // 0 for padded/unused slots
    float* orow = eout + ((size_t)e * CAP + row) * C_DIM + nb + nl;
    orow[0]  = a0[r] * w;
    orow[16] = a1[r] * w;
    orow[32] = a2[r] * w;
    orow[48] = a3[r] * w;
  }
}

// ---------------- 8) deterministic combine: y[t] = sum over kept slots ----------------
__global__ void moe_gather_kernel(const float* __restrict__ eout,
                                  const int* __restrict__ eid,
                                  const int* __restrict__ slotpos,
                                  float* __restrict__ y) {
  size_t i = (size_t)blockIdx.x * 256 + threadIdx.x;
  int t = (int)(i / C_DIM), c = (int)(i % C_DIM);
  float acc = 0.f;
  #pragma unroll
  for (int k = 0; k < 2; ++k) {
    int s = 2 * t + k;
    int p = slotpos[s];
    if (p >= 0) acc += eout[((size_t)eid[s] * CAP + p) * C_DIM + c];
  }
  y[i] = acc;
  if (i == 0) y[(size_t)T_TOK * C_DIM] = 0.f;  // lb_loss (eval mode = 0)
}

// ---------------- launcher ----------------
extern "C" void kernel_launch(void* const* d_in, const int* in_sizes, int n_in,
                              void* d_out, int out_size, void* d_ws, size_t ws_size,
                              hipStream_t stream) {
  const float* x   = (const float*)d_in[0];
  // d_in[1] = t (unused: only selects shared expert bank)
  const float* g1w = (const float*)d_in[2];
  const float* g1b = (const float*)d_in[3];
  const float* g2w = (const float*)d_in[4];
  const float* w1  = (const float*)d_in[5];
  const float* w2  = (const float*)d_in[6];
  float* y = (float*)d_out;

  char* p = (char*)d_ws;
  auto take = [&](size_t bytes) -> char* {
    char* q = p;
    p += (bytes + 255) & ~(size_t)255;
    return q;
  };
  __bf16* gin   = (__bf16*)take((size_t)T_TOK * GIN_LD * 2);
  __bf16* g1wb  = (__bf16*)take((size_t)H_DIM * GIN_LD * 2);
  __bf16* hbuf  = (__bf16*)take((size_t)T_TOK * H_DIM * 2);
  int*    eidb  = (int*)take((size_t)2 * T_TOK * 4);
  float*  ewtb  = (float*)take((size_t)2 * T_TOK * 4);
  int*    spos  = (int*)take((size_t)2 * T_TOK * 4);
  int*    tokix = (int*)take((size_t)NE * CAP * 4);
  float*  swt   = (float*)take((size_t)NE * CAP * 4);
  int*    cnt   = (int*)take((size_t)NE * 4);
  __bf16* w1b   = (__bf16*)take((size_t)NE * 2 * IPAD * C_DIM * 2);
  __bf16* w2b   = (__bf16*)take((size_t)NE * C_DIM * IPAD * 2);
  __bf16* actb  = (__bf16*)take((size_t)NE * CAP * IPAD * 2);
  float*  eout  = (float*)take((size_t)NE * CAP * C_DIM * 4);

  // stats + gate
  stats_gin_kernel<<<T_TOK / 8, 256, 0, stream>>>(x, gin);
  conv_g1w_kernel<<<(int)(((size_t)H_DIM * GIN_LD + 255) / 256), 256, 0, stream>>>(g1w, g1wb);
  gate_gemm_kernel<<<1024, 256, 0, stream>>>(gin, g1wb, g1b, hbuf);
  gate2_route_kernel<<<T_TOK / 8, 256, 0, stream>>>(hbuf, g2w, eidb, ewtb);
  route_compact_kernel<<<NE, 1024, 0, stream>>>(eidb, ewtb, tokix, swt, spos, cnt);

  // expert weight repack (fp32 -> bf16, padded/rearranged)
  conv_w1_kernel<<<(int)(((size_t)NE * 2 * IPAD * C_DIM + 255) / 256), 256, 0, stream>>>(w1, w1b);
  conv_w2_kernel<<<(int)(((size_t)NE * C_DIM * IPAD + 255) / 256), 256, 0, stream>>>(w2, w2b);

  // expert GEMMs (block-cooperative LDS-staged B, 8 waves/block)
  expert_up_kernel<<<NE * (MT_CAP / 8) * (IPAD / 32), 256, 0, stream>>>(gin, w1b, tokix, actb);
  expert_down_kernel<<<NE * (MT_CAP / 8) * (C_DIM / 64), 256, 0, stream>>>(actb, w2b, swt, eout);

  // combine
  moe_gather_kernel<<<(int)(((size_t)T_TOK * C_DIM) / 256), 256, 0, stream>>>(eout, eidb, spos, y);
}
